// GlobalAttention_36361193128044
// MI455X (gfx1250) — matrix operand order repaired and checked by reading, not compiled, not run on hardware
//
#include <hip/hip_runtime.h>

// ---------------- types ----------------
typedef __attribute__((ext_vector_type(16))) __bf16 v16bf;
typedef __attribute__((ext_vector_type(8)))  float  v8f;
typedef __attribute__((ext_vector_type(4)))  unsigned int u32x4;
typedef __attribute__((ext_vector_type(4)))  int    i32x4;
typedef __attribute__((ext_vector_type(8)))  int    i32x8;

union BF16Frag {
    v16bf        v;
    unsigned int u[8];
};

#define NB   4
#define CCH  256
#define HWN  4096

// cheap f32 -> bf16 (round-to-nearest, ties-away)
__device__ inline unsigned short f32_to_bf16(float f) {
    return (unsigned short)((__float_as_uint(f) + 0x8000u) >> 16);
}
// pack two f32 -> bf16x2 in 3 VALU (2x v_add_u32 + v_perm_b32)
__device__ inline unsigned int pack_bf16x2(float lo, float hi) {
    unsigned a = __float_as_uint(lo) + 0x8000u;
    unsigned b = __float_as_uint(hi) + 0x8000u;
    return __builtin_amdgcn_perm(b, a, 0x07060302u);  // {b[31:16], a[31:16]}
}

__device__ inline v8f wmma_bf16(const BF16Frag& A, const BF16Frag& B, v8f C) {
    return __builtin_amdgcn_wmma_f32_16x16x32_bf16(
        false, A.v, false, B.v, (short)0, C, false, false);
}

__device__ inline void load_frag_g(BF16Frag& F, const unsigned short* p) {
    uint4 r0 = *(const uint4*)p;
    uint4 r1 = *(const uint4*)(p + 8);
    F.u[0] = r0.x; F.u[1] = r0.y; F.u[2] = r0.z; F.u[3] = r0.w;
    F.u[4] = r1.x; F.u[5] = r1.y; F.u[6] = r1.z; F.u[7] = r1.w;
}

// ---------------- Tensor Data Mover ----------------
#if defined(__has_builtin)
#if __has_builtin(__builtin_amdgcn_tensor_load_to_lds) && \
    __has_builtin(__builtin_amdgcn_s_wait_tensorcnt)
#define HAVE_TDM 1
#endif
#endif

#ifdef HAVE_TDM
// 2D tile load Global->LDS via TDM (D# groups 0/1 per CDNA5 ISA 8.3/8.4).
// padcfg ORs pad_enable/pad_interval/pad_amount into group1 word0.
__device__ inline void tdm_load_tile(unsigned lds_off, const void* gptr,
                                     unsigned tile0, unsigned tile1,
                                     unsigned dim0, unsigned dim1,
                                     unsigned long long stride0,
                                     unsigned padcfg) {
    unsigned long long ga = (unsigned long long)(uintptr_t)gptr;
    u32x4 g0;
    g0[0] = 1u;                                   // count=1, user descriptor
    g0[1] = lds_off;                              // lds_addr [63:32]
    g0[2] = (unsigned)ga;                         // global_addr lo
    g0[3] = (unsigned)((ga >> 32) & 0x01FFFFFFu) | (2u << 30);  // +type=2
    i32x8 g1;
    g1[0] = (int)((1u << 16) | padcfg);           // wg_mask=0, data_size=2B
    g1[1] = (int)((dim0 & 0xFFFFu) << 16);        // tensor_dim0 @48
    g1[2] = (int)((dim0 >> 16) | ((dim1 & 0xFFFFu) << 16));
    g1[3] = (int)((dim1 >> 16) | (tile0 << 16));  // tile_dim0 @112
    g1[4] = (int)(tile1 & 0xFFFFu);               // tile_dim1; tile_dim2=0
    g1[5] = (int)(stride0 & 0xFFFFFFFFull);       // dim0_stride @160
    g1[6] = (int)((stride0 >> 32) & 0xFFFFull);
    g1[7] = 0;
    i32x4 z4 = {0, 0, 0, 0};
#if __clang_major__ >= 23
    i32x8 z8 = {0, 0, 0, 0, 0, 0, 0, 0};
    __builtin_amdgcn_tensor_load_to_lds(g0, g1, z4, z4, z8, 0);
#else
    __builtin_amdgcn_tensor_load_to_lds(g0, g1, z4, z4, 0);
#endif
}
// pad 4 dwords (16B) after every 16 dwords (64B = one 32-elem bf16 row)
#define PAD_K ((1u << 20) | (3u << 22) | (3u << 25))
// pad 4 dwords (16B) after every 128 dwords (512B = one 256-elem bf16 row)
#define PAD_V ((1u << 20) | (6u << 22) | (3u << 25))
#endif

// padded LDS row strides (halfwords)
#define KSTR 40    // 32 elems + 8 pad  -> 80B rows, bank stride 20 (2-way)
#define VSTR 264   // 256 elems + 8 pad -> 528B rows, bank stride 132 (2-way)

// =====================================================================
// Kernel 1: scaled weight standardization -> bf16 weights (one wave/row)
// =====================================================================
__global__ __launch_bounds__(256) void ws_norm_kernel(
    const float* __restrict__ wq, const float* __restrict__ gq,
    const float* __restrict__ wk, const float* __restrict__ gk,
    const float* __restrict__ wv, const float* __restrict__ gv,
    unsigned short* __restrict__ wn)
{
    int lane = threadIdx.x & 31;
    int gw   = (blockIdx.x * blockDim.x + threadIdx.x) >> 5;
    int s    = gw >> 8;
    int o    = gw & 255;
    const float* w = (s == 0) ? wq : (s == 1) ? wk : wv;
    const float* g = (s == 0) ? gq : (s == 1) ? gk : gv;

    float vals[8];
    float sum = 0.f, sq = 0.f;
#pragma unroll
    for (int t = 0; t < 8; ++t) {
        float x = w[o * CCH + t * 32 + lane];
        vals[t] = x; sum += x; sq += x * x;
    }
#pragma unroll
    for (int d = 1; d < 32; d <<= 1) {
        sum += __shfl_xor(sum, d);
        sq  += __shfl_xor(sq,  d);
    }
    float mean  = sum * (1.0f / 256.0f);
    float var   = (sq - sum * mean) * (1.0f / 255.0f);      // ddof=1
    float scale = g[o] * rsqrtf(fmaxf(var * 256.0f, 1e-4f));
    float shift = mean * scale;
#pragma unroll
    for (int t = 0; t < 8; ++t)
        wn[((size_t)s * CCH + o) * CCH + t * 32 + lane] =
            f32_to_bf16(vals[t] * scale - shift);
}

// =====================================================================
// Kernel 1b: one-shot fp32 -> bf16 conversion of activations a, b
// =====================================================================
__global__ __launch_bounds__(256) void cvt_bf16_kernel(
    const float* __restrict__ a, const float* __restrict__ b,
    unsigned short* __restrict__ aB, unsigned short* __restrict__ bB)
{
    const size_t NV  = (size_t)NB * CCH * HWN / 8;
    size_t tid = (size_t)blockIdx.x * 256 + threadIdx.x;
    const float*    src = (tid < NV) ? a  : b;
    unsigned short* dst = (tid < NV) ? aB : bB;
    size_t off = ((tid < NV) ? tid : tid - NV) * 8;
    float4 f0 = *(const float4*)(src + off);
    float4 f1 = *(const float4*)(src + off + 4);
    uint4 o;
    o.x = pack_bf16x2(f0.x, f0.y);
    o.y = pack_bf16x2(f0.z, f0.w);
    o.z = pack_bf16x2(f1.x, f1.y);
    o.w = pack_bf16x2(f1.z, f1.w);
    *(uint4*)(dst + off) = o;
}

// =====================================================================
// Kernel 2: q/k/v 1x1-conv GEMM (bf16 WMMA, f32 accum)
// =====================================================================
__global__ __launch_bounds__(256) void qkv_kernel(
    const unsigned short* __restrict__ aB, const unsigned short* __restrict__ bB,
    const unsigned short* __restrict__ wn,
    const float* __restrict__ bq, const float* __restrict__ bk,
    const float* __restrict__ bv,
    unsigned short* __restrict__ qT, unsigned short* __restrict__ kB,
    unsigned short* __restrict__ vT)
{
    __shared__ unsigned short Wt[16 * CCH];   // 8 KB weight tile

    int x = blockIdx.x;
    int ichunk = x & 31;  x >>= 5;
    int ot     = x & 15;  x >>= 4;
    int s      = x % 3;
    int batch  = x / 3;

    {
        const uint4* src = (const uint4*)(wn + ((size_t)s * CCH + ot * 16) * CCH);
        uint4*       dst = (uint4*)Wt;
        for (int t = threadIdx.x; t < 512; t += 256) dst[t] = src[t];
    }
    __syncthreads();

    int wave  = threadIdx.x >> 5;
    int lane  = threadIdx.x & 31;
    int row   = lane & 15;
    int hbase = (lane < 16) ? 0 : 8;
    int i0    = ichunk * 128 + wave * 16;

    const unsigned short* src = (s == 1) ? bB : aB;
    v8f acc = {};

#pragma unroll
    for (int kk = 0; kk < 8; ++kk) {
        BF16Frag A;   // 16x32: lane row, K runs {hb..hb+7} U {hb+16..hb+23}
        {
            const unsigned short* p = &Wt[row * CCH + kk * 32 + hbase];
            uint4 r0 = *(const uint4*)p;
            uint4 r1 = *(const uint4*)(p + 16);
            A.u[0] = r0.x; A.u[1] = r0.y; A.u[2] = r0.z; A.u[3] = r0.w;
            A.u[4] = r1.x; A.u[5] = r1.y; A.u[6] = r1.z; A.u[7] = r1.w;
        }
        BF16Frag Bf;  // 32x16: lane col n, K runs {hb..hb+7} U {hb+16..hb+23}
        {
            const unsigned short* bp = src + ((size_t)batch * CCH + kk * 32 + hbase) * HWN + i0 + row;
#pragma unroll
            for (int e = 0; e < 8; ++e) {
                int kq = (e < 4) ? 2 * e : 16 + 2 * (e - 4);
                Bf.u[e] = (unsigned)bp[(size_t)kq * HWN] | ((unsigned)bp[(size_t)(kq + 1) * HWN] << 16);
            }
        }
        acc = wmma_bf16(A, Bf, acc);
    }

    const float* bias = (s == 0) ? bq : (s == 1) ? bk : bv;
#pragma unroll
    for (int r = 0; r < 8; ++r) {
        int o    = ot * 16 + r + hbase;
        int icol = i0 + row;
        unsigned short hv = f32_to_bf16(acc[r] + bias[o]);
        if (s == 1) {
            kB[((size_t)batch * CCH + o) * HWN + icol] = hv;   // [c][j]
        } else {
            unsigned short* dst = (s == 0) ? qT : vT;
            dst[((size_t)batch * HWN + icol) * CCH + o] = hv;  // [i][c]
        }
    }
}

// =====================================================================
// Kernel 3: flash attention, TDM-staged (padded) k/v tiles, mask + blend
// grid = B*64; block = 4 waves (128 thr); wave owns 16 query rows;
// j-tile = 32 keys. Dynamic LDS: 2x(k 20KB + v 16.5KB) + P 4KB.
// =====================================================================
#define KSZ   (256u * KSTR * 2u)            // 20480 B
#define VSZ   (32u * VSTR * 2u)             // 16896 B
#define BUFSZ (KSZ + VSZ)                   // 37376 B
#define KOFF(buf) ((unsigned)(buf) * BUFSZ)
#define VOFF(buf) ((unsigned)(buf) * BUFSZ + KSZ)
#define POFF      (2u * BUFSZ)
#define SMEM_BYTES (POFF + 4096u)

__device__ inline void issue_tiles(char* smem, int buf, int batch, int j0,
                                   const unsigned short* kB,
                                   const unsigned short* vT,
                                   int wave, int tid)
{
#ifdef HAVE_TDM
    if (wave == 0) {
        // k tile: 256 rows x 32 cols (row stride HWN), +16B LDS pad per row
        tdm_load_tile(KOFF(buf) + (unsigned)(uintptr_t)smem,
                      kB + ((size_t)batch * CCH) * HWN + j0,
                      /*tile*/32, 256, /*tensor*/HWN, CCH, /*stride0*/HWN,
                      PAD_K);
        // v tile: 32*256 contiguous elements, +16B LDS pad per 512B row
        tdm_load_tile(VOFF(buf) + (unsigned)(uintptr_t)smem,
                      vT + ((size_t)batch * HWN + j0) * CCH,
                      /*tile*/8192, 0, /*tensor*/1u << 20, 1, 0,
                      PAD_V);
    }
#else
    // fallback: cooperative synchronous copy into the same padded layout
    unsigned short* kd = (unsigned short*)(smem + KOFF(buf));
    unsigned short* vd = (unsigned short*)(smem + VOFF(buf));
    const unsigned short* ks = kB + ((size_t)batch * CCH) * HWN + j0;
    const unsigned short* vs = vT + ((size_t)batch * HWN + j0) * CCH;
    for (int t = tid; t < 1024; t += 128) {
        int r = t >> 2, c4 = t & 3;
        *(uint4*)(kd + r * KSTR + c4 * 8) =
            *(const uint4*)(ks + (size_t)r * HWN + c4 * 8);
    }
    for (int t = tid; t < 1024; t += 128) {
        int r = t >> 5, c8 = t & 31;
        *(uint4*)(vd + r * VSTR + c8 * 8) =
            *(const uint4*)(vs + (size_t)r * CCH + c8 * 8);
    }
#endif
}

__global__ __launch_bounds__(128) void attn_kernel(
    const float* __restrict__ a_in, const float* __restrict__ c_in,
    const unsigned short* __restrict__ qT, const unsigned short* __restrict__ kB,
    const unsigned short* __restrict__ vT, const float* __restrict__ gamma,
    float* __restrict__ out)
{
    extern __shared__ char smem[];

    int batch = blockIdx.x >> 6;
    int iblk  = blockIdx.x & 63;
    int wave  = threadIdx.x >> 5;
    int lane  = threadIdx.x & 31;
    int row   = lane & 15;
    int hbase = (lane < 16) ? 0 : 8;
    int i0    = iblk * 64 + wave * 16;
    unsigned short* Pw = (unsigned short*)(smem + POFF) + wave * 512;

    // ---- preload Q A-fragments (16 rows x K=256) ----
    BF16Frag Qf[8];
    {
        const unsigned short* qrow = qT + ((size_t)batch * HWN + i0 + row) * CCH;
#pragma unroll
        for (int kk = 0; kk < 8; ++kk) {
            uint4 r0 = *(const uint4*)(qrow + kk * 32 + hbase);
            uint4 r1 = *(const uint4*)(qrow + kk * 32 + hbase + 16);
            Qf[kk].u[0] = r0.x; Qf[kk].u[1] = r0.y; Qf[kk].u[2] = r0.z; Qf[kk].u[3] = r0.w;
            Qf[kk].u[4] = r1.x; Qf[kk].u[5] = r1.y; Qf[kk].u[6] = r1.z; Qf[kk].u[7] = r1.w;
        }
    }

    float mi[8], mrun[8], lrun[8];
#pragma unroll
    for (int r = 0; r < 8; ++r) {
        int i = i0 + r + hbase;
        int h = i >> 6, w = i & 63;
        mi[r]   = c_in[batch * 1024 + (h >> 1) * 32 + (w >> 1)];
        mrun[r] = -3.0e38f;
        lrun[r] = 0.0f;
    }

    v8f zf = {};
    v8f O[16];
#pragma unroll
    for (int t = 0; t < 16; ++t) O[t] = zf;

    int buf = 0;
    issue_tiles(smem, 0, batch, 0, kB, vT, wave, threadIdx.x);

    for (int jt = 0; jt < HWN / 32; ++jt) {
#ifdef HAVE_TDM
        __builtin_amdgcn_s_wait_tensorcnt(0);  // no-op for non-issuing waves
#endif
        __syncthreads();                       // tile[buf] visible to all
        if (jt + 1 < HWN / 32)
            issue_tiles(smem, buf ^ 1, batch, (jt + 1) * 32, kB, vT,
                        wave, threadIdx.x);

        const unsigned short* kLds = (const unsigned short*)(smem + KOFF(buf));
        const unsigned short* vLds = (const unsigned short*)(smem + VOFF(buf));

        // ---- S = q^T k : two 16x16 tiles over K=256 ----
        v8f S0 = zf, S1 = zf;
#pragma unroll
        for (int kk = 0; kk < 8; ++kk) {
            const unsigned short* kp = kLds + (kk * 32 + hbase) * KSTR + row;
            BF16Frag B0, B1;
            {
#pragma unroll
                for (int e = 0; e < 8; ++e) {
                    int kq = (e < 4) ? 2 * e : 16 + 2 * (e - 4);
                    B0.u[e] = (unsigned)kp[kq * KSTR] | ((unsigned)kp[(kq + 1) * KSTR] << 16);
                    B1.u[e] = (unsigned)kp[kq * KSTR + 16] | ((unsigned)kp[(kq + 1) * KSTR + 16] << 16);
                }
            }
            S0 = wmma_bf16(Qf[kk], B0, S0);
            S1 = wmma_bf16(Qf[kk], B1, S1);
        }

        // ---- masked online softmax (rows half-lane aligned) ----
#pragma unroll
        for (int r = 0; r < 8; ++r) {
            float s0 = S0[r] * mi[r];
            float s1 = S1[r] * mi[r];
            float rm = fmaxf(s0, s1);
#pragma unroll
            for (int d = 1; d < 16; d <<= 1) rm = fmaxf(rm, __shfl_xor(rm, d));
            float nm   = fmaxf(mrun[r], rm);
            float corr = __expf(mrun[r] - nm);
            float p0   = __expf(s0 - nm);
            float p1   = __expf(s1 - nm);
            float rs   = p0 + p1;
#pragma unroll
            for (int d = 1; d < 16; d <<= 1) rs += __shfl_xor(rs, d);
            lrun[r] = lrun[r] * corr + rs;
            mrun[r] = nm;
#pragma unroll
            for (int t = 0; t < 16; ++t) O[t][r] *= corr;

            int prow = r + hbase;
            Pw[prow * 32 + row]      = f32_to_bf16(p0);
            Pw[prow * 32 + 16 + row] = f32_to_bf16(p1);
        }

        // ---- re-stripe P (C-layout -> A-layout) through LDS ----
        BF16Frag Pf;
        {
            const unsigned short* pp = Pw + row * 32 + hbase;
            uint4 r0 = *(const uint4*)pp;
            uint4 r1 = *(const uint4*)(pp + 16);
            Pf.u[0] = r0.x; Pf.u[1] = r0.y; Pf.u[2] = r0.z; Pf.u[3] = r0.w;
            Pf.u[4] = r1.x; Pf.u[5] = r1.y; Pf.u[6] = r1.z; Pf.u[7] = r1.w;
        }

        // ---- O += P x v^T : 16 channel tiles, K = 32 keys ----
#pragma unroll
        for (int t = 0; t < 16; ++t) {
            BF16Frag Bv;
            {
                const unsigned short* vp = vLds + hbase * VSTR + t * 16 + row;
#pragma unroll
                for (int e = 0; e < 8; ++e) {
                    int kq = (e < 4) ? 2 * e : 16 + 2 * (e - 4);
                    Bv.u[e] = (unsigned)vp[kq * VSTR] | ((unsigned)vp[(kq + 1) * VSTR] << 16);
                }
            }
            O[t] = wmma_bf16(Pf, Bv, O[t]);
        }

        __syncthreads();                       // done with tile[buf]
        buf ^= 1;
    }

    // ---- epilogue: normalize, mask-blend, store ----
    float g = gamma[0];
#pragma unroll
    for (int r = 0; r < 8; ++r) {
        int   i   = i0 + r + hbase;
        float inv = 1.0f / lrun[r];
        float m   = mi[r];
#pragma unroll
        for (int t = 0; t < 16; ++t) {
            int    ccol = t * 16 + row;
            size_t idx  = ((size_t)batch * CCH + ccol) * HWN + i;
            out[idx] = a_in[idx] * m + g * (1.0f - m) * (O[t][r] * inv);
        }
    }
}

// =====================================================================
extern "C" void kernel_launch(void* const* d_in, const int* in_sizes, int n_in,
                              void* d_out, int out_size, void* d_ws, size_t ws_size,
                              hipStream_t stream) {
    const float* a     = (const float*)d_in[0];
    const float* b     = (const float*)d_in[1];
    const float* c     = (const float*)d_in[2];
    const float* wq    = (const float*)d_in[3];
    const float* bq    = (const float*)d_in[4];
    const float* gq    = (const float*)d_in[5];
    const float* wk    = (const float*)d_in[6];
    const float* bk    = (const float*)d_in[7];
    const float* gk    = (const float*)d_in[8];
    const float* wv    = (const float*)d_in[9];
    const float* bv    = (const float*)d_in[10];
    const float* gv    = (const float*)d_in[11];
    const float* gamma = (const float*)d_in[12];
    float* out = (float*)d_out;

    char* ws = (char*)d_ws;
    const size_t WN_BYTES  = (size_t)3 * CCH * CCH * 2;        // 384 KB
    const size_t MAT_BYTES = (size_t)NB * HWN * CCH * 2;       // 8 MB each
    unsigned short* wn = (unsigned short*)ws;
    unsigned short* aB = (unsigned short*)(ws + WN_BYTES);
    unsigned short* bB = (unsigned short*)(ws + WN_BYTES + MAT_BYTES);
    unsigned short* qT = (unsigned short*)(ws + WN_BYTES + 2 * MAT_BYTES);
    unsigned short* kB = (unsigned short*)(ws + WN_BYTES + 3 * MAT_BYTES);
    unsigned short* vT = (unsigned short*)(ws + WN_BYTES + 4 * MAT_BYTES);

    ws_norm_kernel<<<96, 256, 0, stream>>>(wq, gq, wk, gk, wv, gv, wn);
    cvt_bf16_kernel<<<4096, 256, 0, stream>>>(a, b, aB, bB);
    qkv_kernel<<<NB * 3 * 16 * 32, 256, 0, stream>>>(aB, bB, wn, bq, bk, bv, qT, kB, vT);
    attn_kernel<<<NB * 64, 128, SMEM_BYTES, stream>>>(a, c, qT, kB, vT, gamma, out);
}
